// AgentAttention_18846316494961
// MI455X (gfx1250) — matrix operand order repaired and checked by reading, not compiled, not run on hardware
//
#include <hip/hip_runtime.h>
#include <hip/hip_bf16.h>

typedef _Float16 half_t;
typedef __attribute__((ext_vector_type(16))) _Float16 v16h;
typedef __attribute__((ext_vector_type(8)))  _Float16 v8h;
typedef __attribute__((ext_vector_type(8)))  float    v8f;

#define NB     16
#define CCH    512
#define NHEAD  16
#define HD     32
#define NTOK   4096
#define TTOT   (NB * NTOK)
#define AGENTS 49
#define SCALE  0.17677669529663689f  /* 1/sqrt(32) */

// ---------------------------------------------------------------------------
// WMMA helpers (CDNA5 wave32, V_WMMA_F32_16X16X32_F16)
// ---------------------------------------------------------------------------
__device__ __forceinline__ v8f wmma16(v16h a, v16h b, v8f c) {
  return __builtin_amdgcn_wmma_f32_16x16x32_f16(false, a, false, b, (short)0, c,
                                                false, false);
}

// A fragment: 16x32 f16, row-major source, row = lane%16.
// lane<16 holds K {0..7,16..23}; lane>=16 holds K {8..15,24..31}.
__device__ __forceinline__ v16h frag_a(const half_t* p, int lda) {
  int lane = threadIdx.x & 31;
  const half_t* r = p + (size_t)(lane & 15) * lda + ((lane >> 4) << 3);
  v8h lo = *(const v8h*)r;
  v8h hi = *(const v8h*)(r + 16);
  v16h a;
#pragma unroll
  for (int i = 0; i < 8; ++i) { a[i] = lo[i]; a[i + 8] = hi[i]; }
  return a;
}

// B fragment: 32x16 f16 given Bt (N x K row-major). Lane holds column
// n = lane%16; lane<16 holds K {0..7,16..23}; lane>=16 holds K {8..15,24..31}.
__device__ __forceinline__ v16h frag_b(const half_t* p, int ldb) {
  int lane = threadIdx.x & 31;
  const half_t* r = p + (size_t)(lane & 15) * ldb + ((lane >> 4) << 3);
  v8h lo = *(const v8h*)r;
  v8h hi = *(const v8h*)(r + 16);
  v16h b;
#pragma unroll
  for (int i = 0; i < 8; ++i) { b[i] = lo[i]; b[i + 8] = hi[i]; }
  return b;
}

// ---------------------------------------------------------------------------
// 1) conv 1x1 input projection -> xf f16, token order t = w*64 + h
// ---------------------------------------------------------------------------
__global__ void k_conv_in(const float* __restrict__ x, const float* __restrict__ in_w,
                          const float* __restrict__ in_b, half_t* __restrict__ xf) {
  int gt = blockIdx.x;            // b*4096 + t
  int b = gt >> 12, t = gt & 4095;
  int h = t & 63, w = t >> 6;
  float x0 = x[(size_t)(b * 3 + 0) * 4096 + h * 64 + w];
  float x1 = x[(size_t)(b * 3 + 1) * 4096 + h * 64 + w];
  float x2 = x[(size_t)(b * 3 + 2) * 4096 + h * 64 + w];
  for (int c = threadIdx.x; c < CCH; c += 128) {
    float v = x0 * in_w[c] + x1 * in_w[512 + c] + x2 * in_w[1024 + c] + in_b[c];
    xf[(size_t)gt * CCH + c] = (half_t)v;
  }
}

// ---------------------------------------------------------------------------
// 2) weight transpose + f32->f16:  wt[c][r] = w[r][c]
// ---------------------------------------------------------------------------
__global__ void k_wt(const float* __restrict__ w, half_t* __restrict__ wt,
                     int R, int Ccols) {
  int idx = blockIdx.x * 256 + threadIdx.x;
  if (idx >= R * Ccols) return;
  int r = idx / Ccols, c = idx % Ccols;
  wt[(size_t)c * R + r] = (half_t)w[idx];
}

// ---------------------------------------------------------------------------
// 3) generic WMMA GEMM: C[MxN] = A[MxK] * Bt[NxK]^T + bias
//    block = 256 threads (8 waves); tile 128(M) x 64(N); wave does 32x32.
//    MODE 0: store f16 (ldc).  MODE 1: projection epilogue -> f32 into
//            (B, C, H, W) layout; j-consecutive stores are contiguous in t.
// ---------------------------------------------------------------------------
template <int MODE>
__global__ __launch_bounds__(256) void k_gemm(
    const half_t* __restrict__ A, const half_t* __restrict__ Bt,
    const float* __restrict__ bias, half_t* __restrict__ out16,
    float* __restrict__ out32, int M, int N, int K, int lda, int ldb, int ldc) {
  int wv = threadIdx.x >> 5;
  int wm = wv & 3, wn = wv >> 2;
  int row0 = blockIdx.y * 128 + wm * 32;
  int col0 = blockIdx.x * 64 + wn * 32;
  v8f acc[2][2] = {};
  for (int k0 = 0; k0 < K; k0 += 32) {
    v16h af[2], bf[2];
#pragma unroll
    for (int r = 0; r < 2; ++r)
      af[r] = frag_a(A + (size_t)(row0 + r * 16) * lda + k0, lda);
#pragma unroll
    for (int c = 0; c < 2; ++c)
      bf[c] = frag_b(Bt + (size_t)(col0 + c * 16) * ldb + k0, ldb);
#pragma unroll
    for (int r = 0; r < 2; ++r)
#pragma unroll
      for (int c = 0; c < 2; ++c) acc[r][c] = wmma16(af[r], bf[c], acc[r][c]);
  }
  int lane = threadIdx.x & 31;
#pragma unroll
  for (int r = 0; r < 2; ++r)
#pragma unroll
    for (int c = 0; c < 2; ++c) {
      int n = col0 + c * 16 + (lane & 15);
      float bv = bias ? bias[n] : 0.0f;
      int mbase = row0 + r * 16 + ((lane >> 4) << 3);
#pragma unroll
      for (int j = 0; j < 8; ++j) {
        int m = mbase + j;
        float v = acc[r][c][j] + bv;
        if (MODE == 0) {
          out16[(size_t)m * ldc + n] = (half_t)v;
        } else {
          int b = m >> 12, t = m & 4095;  // contiguous in t across j
          out32[(size_t)(b * CCH + n) * 4096 + t] = v;
        }
      }
    }
}

// ---------------------------------------------------------------------------
// 4) adaptive avg pool q (64x64 -> 7x7, all windows 10x10); qt padded to 64
//    agent rows (rows 49..63 zero).
// ---------------------------------------------------------------------------
__global__ void k_pool(const half_t* __restrict__ qkv, half_t* __restrict__ qt) {
  int b = blockIdx.x >> 6;
  int a = blockIdx.x & 63;
  if (a >= AGENTS) {
    for (int c = threadIdx.x; c < CCH; c += 256)
      qt[((size_t)(b * 64 + a)) * CCH + c] = (half_t)0.0f;
    return;
  }
  int p = a / 7, r = a % 7;
  int si = (p * 64) / 7;
  int sj = (r * 64) / 7;
  for (int c = threadIdx.x; c < CCH; c += 256) {
    float s = 0.0f;
    for (int di = 0; di < 10; ++di)
      for (int dj = 0; dj < 10; ++dj) {
        int n = (si + di) * 64 + (sj + dj);
        s += (float)qkv[((size_t)(b * NTOK + n)) * 1536 + c];
      }
    qt[((size_t)(b * 64 + a)) * CCH + c] = (half_t)(s * 0.01f);
  }
}

// ---------------------------------------------------------------------------
// 5) position biases (bilinear resize 7x7 -> 64x64, align_corners=False)
// ---------------------------------------------------------------------------
__device__ __forceinline__ float bilin7(const float* __restrict__ m, int yi, int xi) {
  float cy = (yi + 0.5f) * (7.0f / 64.0f) - 0.5f;
  float cx = (xi + 0.5f) * (7.0f / 64.0f) - 0.5f;
  cy = fminf(fmaxf(cy, 0.0f), 6.0f);
  cx = fminf(fmaxf(cx, 0.0f), 6.0f);
  int y0 = (int)floorf(cy); int y1 = y0 + 1 > 6 ? 6 : y0 + 1; float wy = cy - (float)y0;
  int x0 = (int)floorf(cx); int x1 = x0 + 1 > 6 ? 6 : x0 + 1; float wx = cx - (float)x0;
  float a0 = m[y0 * 7 + x0] * (1.0f - wy) + m[y1 * 7 + x0] * wy;
  float a1 = m[y0 * 7 + x1] * (1.0f - wy) + m[y1 * 7 + x1] * wy;
  return a0 * (1.0f - wx) + a1 * wx;
}

__global__ void k_bias1(const float* __restrict__ an, const float* __restrict__ ah,
                        const float* __restrict__ aw, float* __restrict__ bias1) {
  int idx = blockIdx.x * 256 + threadIdx.x;  // head*49*4096 + a*4096 + n
  if (idx >= NHEAD * AGENTS * NTOK) return;
  int n = idx & 4095;
  int ha = idx >> 12;
  int a = ha % AGENTS, head = ha / AGENTS;
  int y = n >> 6, x = n & 63;
  const float* m = an + (size_t)(head * AGENTS + a) * 49;
  bias1[idx] = bilin7(m, y, x) + ah[(size_t)(head * AGENTS + a) * 64 + y] +
               aw[(size_t)(head * AGENTS + a) * 64 + x];
}

__global__ void k_bias2(const float* __restrict__ na, const float* __restrict__ ha,
                        const float* __restrict__ wa, float* __restrict__ bias2) {
  int idx = blockIdx.x * 256 + threadIdx.x;  // head*4096*49 + n*49 + a
  if (idx >= NHEAD * NTOK * AGENTS) return;
  int a = idx % AGENTS;
  int rest = idx / AGENTS;
  int n = rest & 4095, head = rest >> 12;
  int y = n >> 6, x = n & 63;
  const float* m = na + (size_t)(head * AGENTS + a) * 49;
  bias2[idx] = bilin7(m, y, x) + ha[((size_t)head * 64 + y) * AGENTS + a] +
               wa[((size_t)head * 64 + x) * AGENTS + a];
}

// ---------------------------------------------------------------------------
// 6) per-channel transpose of v:  vT[b][c][t] = qkv[b*4096+t][1024+c]
// ---------------------------------------------------------------------------
__global__ void k_vt(const half_t* __restrict__ qkv, half_t* __restrict__ vT) {
  int idx = blockIdx.x * 256 + threadIdx.x;  // b*512*4096 + c*4096 + t
  if (idx >= NB * CCH * NTOK) return;
  int t = idx & 4095;
  int bc = idx >> 12;
  int c = bc & 511, b = bc >> 9;
  vT[idx] = qkv[((size_t)(b * NTOK + t)) * 1536 + 1024 + c];
}

// ---------------------------------------------------------------------------
// 7) FUSED stage 1 (flash-style): per (b,head) block, 8 waves each own a
//    512-token column stripe. Online softmax over 64 (padded-49) agent rows:
//    S-WMMA -> shfl row max/sum -> rescale O accumulators -> P via LDS
//    layout-bounce -> PV-WMMA. Cross-wave merge via LDS f32 atomics.
//    __expf keeps the inner loop on the TRANS32 v_exp_f32 path so softmax
//    bookkeeping co-executes with the XDL WMMA pipe instead of drowning it.
// ---------------------------------------------------------------------------
__global__ __launch_bounds__(256) void k_s1_fused(
    const half_t* __restrict__ qkv, const half_t* __restrict__ qt,
    const float* __restrict__ bias1, const half_t* __restrict__ vT,
    half_t* __restrict__ avT) {
  __shared__ __align__(16) half_t ptile[8][64][32];  // per-wave P tile (32 KB)
  __shared__ float lmax[8][64];
  __shared__ float lsum[8][64];
  __shared__ float oacc[64][32];
  __shared__ float glds[64];
  int bh = blockIdx.x;
  int b = bh >> 4, head = bh & 15;
  int wv = threadIdx.x >> 5;
  int lane = threadIdx.x & 31;
  int hi = lane >> 4;
  int col = lane & 15;

  for (int i = threadIdx.x; i < 64 * 32; i += 256) (&oacc[0][0])[i] = 0.0f;

  // constant A fragments: qt rows (this head's 32 channels), K=32 single step
  v16h af[4];
#pragma unroll
  for (int mt = 0; mt < 4; ++mt)
    af[mt] = frag_a(qt + ((size_t)(b * 64 + mt * 16)) * CCH + head * HD, CCH);

  float rm[4][8], rl[4][8];
#pragma unroll
  for (int mt = 0; mt < 4; ++mt)
#pragma unroll
    for (int j = 0; j < 8; ++j) { rm[mt][j] = -3.0e38f; rl[mt][j] = 0.0f; }
  v8f o[4][2] = {};
  v8f zacc = {};

  for (int ch = 0; ch < 16; ++ch) {
    int kc = wv * 512 + ch * 32;  // this wave's token chunk
    v16h bf0 = frag_b(qkv + ((size_t)(b * NTOK + kc)) * 1536 + 512 + head * HD, 1536);
    v16h bf1 = frag_b(qkv + ((size_t)(b * NTOK + kc + 16)) * 1536 + 512 + head * HD, 1536);
    v16h bv0 = frag_b(vT + ((size_t)(b * CCH + head * HD)) * NTOK + kc, NTOK);
    v16h bv1 = frag_b(vT + ((size_t)(b * CCH + head * HD + 16)) * NTOK + kc, NTOK);
#pragma unroll
    for (int mt = 0; mt < 4; ++mt) {
      v8f s0 = wmma16(af[mt], bf0, zacc);
      v8f s1 = wmma16(af[mt], bf1, zacc);
      float fct[8];
#pragma unroll
      for (int j = 0; j < 8; ++j) {
        int m = mt * 16 + j + (hi << 3);
        float v0 = s0[j] * SCALE;
        float v1 = s1[j] * SCALE;
        if (m < AGENTS) {
          const float* bp = bias1 + ((size_t)(head * AGENTS + m)) * NTOK + kc + col;
          v0 += bp[0];
          v1 += bp[16];
        }
        float tmax = fmaxf(v0, v1);
#pragma unroll
        for (int s = 1; s < 16; s <<= 1) tmax = fmaxf(tmax, __shfl_xor(tmax, s, 16));
        float nm = fmaxf(rm[mt][j], tmax);
        float f = __expf(rm[mt][j] - nm);   // 0 on first chunk (rm = -inf)
        float p0 = __expf(v0 - nm), p1 = __expf(v1 - nm);
        float ps = p0 + p1;
#pragma unroll
        for (int s = 1; s < 16; s <<= 1) ps += __shfl_xor(ps, s, 16);
        rl[mt][j] = rl[mt][j] * f + ps;
        rm[mt][j] = nm;
        fct[j] = f;
        int row = mt * 16 + j + (hi << 3);
        ptile[wv][row][col]      = (half_t)p0;
        ptile[wv][row][col + 16] = (half_t)p1;
      }
#pragma unroll
      for (int nt = 0; nt < 2; ++nt)
#pragma unroll
        for (int j = 0; j < 8; ++j) o[mt][nt][j] *= fct[j];
      // LDS ops from the same wave are in-order: safe read-after-write
      v16h a2 = frag_a(&ptile[wv][mt * 16][0], 32);
      o[mt][0] = wmma16(a2, bv0, o[mt][0]);
      o[mt][1] = wmma16(a2, bv1, o[mt][1]);
    }
  }

  // publish per-wave stats (row values are uniform across each 16-lane half)
  if (col == 0) {
#pragma unroll
    for (int mt = 0; mt < 4; ++mt)
#pragma unroll
      for (int j = 0; j < 8; ++j) {
        int row = mt * 16 + j + (hi << 3);
        lmax[wv][row] = rm[mt][j];
        lsum[wv][row] = rl[mt][j];
      }
  }
  __syncthreads();

  // cross-wave merge: rescale partials to the global max and accumulate
#pragma unroll
  for (int mt = 0; mt < 4; ++mt)
#pragma unroll
    for (int j = 0; j < 8; ++j) {
      int row = mt * 16 + j + (hi << 3);
      float gm = -3.0e38f;
      for (int w = 0; w < 8; ++w) gm = fmaxf(gm, lmax[w][row]);
      float fw = __expf(rm[mt][j] - gm);
      if (wv == 0 && col == 0) {
        float gl = 0.0f;
        for (int w = 0; w < 8; ++w) gl += lsum[w][row] * __expf(lmax[w][row] - gm);
        glds[row] = gl;
      }
#pragma unroll
      for (int nt = 0; nt < 2; ++nt)
        atomicAdd(&oacc[row][nt * 16 + col], o[mt][nt][j] * fw);
    }
  __syncthreads();

  // normalize and store transposed: avT[bh][d][m]
  for (int i = threadIdx.x; i < 64 * 32; i += 256) {
    int m = i & 63;  // d = i >> 6
    avT[(size_t)bh * 2048 + i] = (half_t)(oacc[m][i >> 6] / glds[m]);
  }
}

// ---------------------------------------------------------------------------
// 8) FUSED stage 2: logits WMMA -> masked softmax over 49 agents (cross-lane
//    shfl within the owning wave) -> LDS layout bounce -> P @ agent_v WMMA.
// ---------------------------------------------------------------------------
__global__ __launch_bounds__(256) void k_s2_fused(
    const half_t* __restrict__ qkv, const half_t* __restrict__ qt,
    const float* __restrict__ bias2, const half_t* __restrict__ avT,
    float* __restrict__ out_attn) {
  __shared__ __align__(16) half_t ptile[8][16][64];  // per-wave P tile, 16 KB
  int bh = blockIdx.y;
  int b = bh >> 4, head = bh & 15;
  int wv = threadIdx.x >> 5;
  int lane = threadIdx.x & 31;
  int hi = lane >> 4;
  int m0 = blockIdx.x * 128 + wv * 16;

  // logits: (16 x 32) @ (32 x 64)
  v16h af = frag_a(qkv + ((size_t)(b * NTOK + m0)) * 1536 + head * HD, 1536);
  v8f acc[4] = {};
#pragma unroll
  for (int nt = 0; nt < 4; ++nt) {
    v16h bf = frag_b(qt + ((size_t)(b * 64 + nt * 16)) * CCH + head * HD, CCH);
    acc[nt] = wmma16(af, bf, acc[nt]);
  }

  // scale + bias + agent mask
  float lg[4][8];
#pragma unroll
  for (int nt = 0; nt < 4; ++nt) {
    int n = nt * 16 + (lane & 15);
#pragma unroll
    for (int j = 0; j < 8; ++j) {
      int m = m0 + j + (hi << 3);
      lg[nt][j] = (n < AGENTS)
          ? acc[nt][j] * SCALE + bias2[((size_t)head * NTOK + m) * AGENTS + n]
          : -3.0e38f;
    }
  }

  // per-row softmax: reduce over 4 regs + 16 lanes of the owning half-wave
#pragma unroll
  for (int j = 0; j < 8; ++j) {
    float mx = fmaxf(fmaxf(lg[0][j], lg[1][j]), fmaxf(lg[2][j], lg[3][j]));
#pragma unroll
    for (int s = 1; s < 16; s <<= 1) mx = fmaxf(mx, __shfl_xor(mx, s, 16));
    float e0 = __expf(lg[0][j] - mx), e1 = __expf(lg[1][j] - mx);
    float e2 = __expf(lg[2][j] - mx), e3 = __expf(lg[3][j] - mx);
    float sum = e0 + e1 + e2 + e3;
#pragma unroll
    for (int s = 1; s < 16; s <<= 1) sum += __shfl_xor(sum, s, 16);
    float inv = 1.0f / sum;
    int row = j + (hi << 3);
    int cl = lane & 15;
    ptile[wv][row][cl]      = (half_t)(e0 * inv);
    ptile[wv][row][cl + 16] = (half_t)(e1 * inv);
    ptile[wv][row][cl + 32] = (half_t)(e2 * inv);
    ptile[wv][row][cl + 48] = (half_t)(e3 * inv);
  }
  __syncthreads();

  // out = P (16 x 64) @ agent_v (64 x 32)
  const half_t* Bt = avT + (size_t)bh * 32 * 64;
  v8f o[2] = {};
#pragma unroll
  for (int k0 = 0; k0 < 64; k0 += 32) {
    v16h a = frag_a(&ptile[wv][0][0] + k0, 64);
#pragma unroll
    for (int nt = 0; nt < 2; ++nt) {
      v16h bb = frag_b(Bt + (size_t)(nt * 16) * 64 + k0, 64);
      o[nt] = wmma16(a, bb, o[nt]);
    }
  }
#pragma unroll
  for (int nt = 0; nt < 2; ++nt) {
    int n = head * HD + nt * 16 + (lane & 15);
#pragma unroll
    for (int j = 0; j < 8; ++j) {
      int m = m0 + j + (hi << 3);
      out_attn[((size_t)(b * NTOK + m)) * CCH + n] = o[nt][j];
    }
  }
}

// ---------------------------------------------------------------------------
// 9) depthwise 3x3 on v + bias + attention residual -> proj_in f16
// ---------------------------------------------------------------------------
__global__ void k_dwc(const half_t* __restrict__ qkv, const float* __restrict__ out_attn,
                      const float* __restrict__ dwc_w, const float* __restrict__ dwc_b,
                      half_t* __restrict__ proj_in) {
  int gt = blockIdx.x;  // b*4096 + n
  int b = gt >> 12, n = gt & 4095;
  int i = n >> 6, j = n & 63;
  for (int c = threadIdx.x; c < CCH; c += 256) {
    float s = dwc_b[c];
    for (int di = -1; di <= 1; ++di)
      for (int dj = -1; dj <= 1; ++dj) {
        int ii = i + di, jj = j + dj;
        if (ii >= 0 && ii < 64 && jj >= 0 && jj < 64)
          s += (float)qkv[((size_t)(b * NTOK + ii * 64 + jj)) * 1536 + 1024 + c] *
               dwc_w[c * 9 + (di + 1) * 3 + (dj + 1)];
      }
    proj_in[(size_t)gt * CCH + c] = (half_t)(out_attn[(size_t)gt * CCH + c] + s);
  }
}

// ---------------------------------------------------------------------------
extern "C" void kernel_launch(void* const* d_in, const int* in_sizes, int n_in,
                              void* d_out, int out_size, void* d_ws, size_t ws_size,
                              hipStream_t stream) {
  (void)in_sizes; (void)n_in; (void)out_size; (void)ws_size;
  const float* x       = (const float*)d_in[0];
  const float* in_w    = (const float*)d_in[1];
  const float* in_b    = (const float*)d_in[2];
  const float* qkv_w   = (const float*)d_in[3];
  const float* qkv_b   = (const float*)d_in[4];
  const float* proj_w  = (const float*)d_in[5];
  const float* proj_b  = (const float*)d_in[6];
  const float* dwc_w   = (const float*)d_in[7];
  const float* dwc_b   = (const float*)d_in[8];
  const float* an_bias = (const float*)d_in[9];
  const float* na_bias = (const float*)d_in[10];
  const float* ah_bias = (const float*)d_in[11];
  const float* aw_bias = (const float*)d_in[12];
  const float* ha_bias = (const float*)d_in[13];
  const float* wa_bias = (const float*)d_in[14];
  float* out = (float*)d_out;

  char* ws = (char*)d_ws;
  size_t off = 0;
  auto alloc = [&](size_t bytes) -> void* {
    void* p = ws + off;
    off = (off + bytes + 255) & ~(size_t)255;
    return p;
  };
  half_t* qkv16 = (half_t*)alloc((size_t)TTOT * 1536 * 2);       // q|k|v f16
  half_t* xf16  = (half_t*)alloc((size_t)TTOT * CCH * 2);        // reused: proj_in
  half_t* wT1   = (half_t*)alloc((size_t)1536 * 512 * 2);
  half_t* wT2   = (half_t*)alloc((size_t)512 * 512 * 2);
  half_t* qt16  = (half_t*)alloc((size_t)NB * 64 * CCH * 2);
  float*  bias1 = (float*)alloc((size_t)NHEAD * AGENTS * NTOK * 4);
  float*  bias2 = (float*)alloc((size_t)NHEAD * NTOK * AGENTS * 4);
  half_t* avT   = (half_t*)alloc((size_t)NB * NHEAD * 32 * 64 * 2);
  half_t* vT    = (half_t*)alloc((size_t)NB * CCH * NTOK * 2);
  float*  out_attn = (float*)alloc((size_t)TTOT * CCH * 4);
  half_t* proj_in  = xf16;   // dead after QKV GEMM

  // 1. input projection
  k_conv_in<<<TTOT, 128, 0, stream>>>(x, in_w, in_b, xf16);
  // 2. weight transposes
  k_wt<<<(512 * 1536 + 255) / 256, 256, 0, stream>>>(qkv_w, wT1, 512, 1536);
  k_wt<<<(512 * 512 + 255) / 256, 256, 0, stream>>>(proj_w, wT2, 512, 512);
  // 3. QKV GEMM (65536 x 1536 x 512), f16 out
  k_gemm<0><<<dim3(1536 / 64, TTOT / 128), 256, 0, stream>>>(
      xf16, wT1, qkv_b, qkv16, nullptr, TTOT, 1536, 512, 512, 512, 1536);
  // 4. agent pooling
  k_pool<<<NB * 64, 256, 0, stream>>>(qkv16, qt16);
  // 5. position biases
  k_bias1<<<(NHEAD * AGENTS * NTOK + 255) / 256, 256, 0, stream>>>(an_bias, ah_bias,
                                                                   aw_bias, bias1);
  k_bias2<<<(NHEAD * NTOK * AGENTS + 255) / 256, 256, 0, stream>>>(na_bias, ha_bias,
                                                                   wa_bias, bias2);
  // 6. v transpose (per-head contiguous K)
  k_vt<<<(NB * CCH * NTOK) / 256, 256, 0, stream>>>(qkv16, vT);
  // 7. stage 1 fully fused (flash-style online softmax, no S1/P1 round trip)
  k_s1_fused<<<NB * NHEAD, 256, 0, stream>>>(qkv16, qt16, bias1, vT, avT);
  // 8. stage 2 fully fused
  k_s2_fused<<<dim3(32, NB * NHEAD), 256, 0, stream>>>(qkv16, qt16, bias2, avT,
                                                       out_attn);
  // 9. depthwise conv residual
  k_dwc<<<TTOT, 256, 0, stream>>>(qkv16, out_attn, dwc_w, dwc_b, proj_in);
  // 10. output projection into (B,C,H,W)
  k_gemm<1><<<dim3(512 / 64, TTOT / 128), 256, 0, stream>>>(
      proj_in, wT2, proj_b, nullptr, out, TTOT, 512, 512, 512, 512, 0);
}
